// Ross_18846316495296
// MI455X (gfx1250) — matrix operand order repaired and checked by reading, not compiled, not run on hardware
//
#include <hip/hip_runtime.h>
#include <hip/hip_bf16.h>

// Problem dims
#define S_  30
#define B_  16
#define T_  8
#define V_  20000
#define H_  400
#define LU_ 256
#define LS_ 64
#define NG_ 3
#define HP  416           // K padded to 13*32 for bf16 WMMA
#define KT  13            // K tiles of 32
#define R_  (S_*B_)       // 480 rows = 30 tiles of 16
#define NVB 157           // vocab blocks of 128 cols
#define VP  (NVB*128)     // 20096 padded vocab rows

// k_vocab dynamic-LDS layout (dynamic-only shared => base offset 0)
#define A_OFF   0u                  // 16 x HP bf16 = 13312 B  (TDM staged)
#define B0_OFF  13312u              // 128 rows x 64 B = 8192 B
#define B1_OFF  21504u
#define LM_OFF  29696u              // 4x16 floats
#define LSM_OFF 29952u              // 4x16 floats
#define VOC_LDS 30208u

typedef __attribute__((ext_vector_type(16))) __bf16 v16bf;
typedef __attribute__((ext_vector_type(8)))  __bf16 v8bf;
typedef __attribute__((ext_vector_type(8)))  float  v8f;
typedef unsigned u32x4 __attribute__((ext_vector_type(4)));
typedef unsigned u32x8 __attribute__((ext_vector_type(8)));

#define NEG_INF (-__builtin_inff())

__device__ __forceinline__ __bf16 f2bf(float f) {
    unsigned u = __float_as_uint(f);
    unsigned r = (u + 0x7fffu + ((u >> 16) & 1u)) >> 16;
    unsigned short h = (unsigned short)r;
    return __builtin_bit_cast(__bf16, h);
}

__device__ __forceinline__ v8f wmma_bf16(v16bf a, v16bf b, v8f c) {
    return __builtin_amdgcn_wmma_f32_16x16x32_bf16(false, a, false, b, (short)0, c, false, false);
}

// A tile (16xK=32) from row-major [M][K] buffer, per CDNA5 A layout:
// lane m / m+16 holds row m; halves i<8 -> K=i+hi*8, i>=8 -> K=(i-8)+16+hi*8
__device__ __forceinline__ v16bf load_a16(const __bf16* A, int lda, int lane) {
    int m = lane & 15, hi = (lane >> 4) & 1;
    const __bf16* p = A + (size_t)m * lda + hi * 8;
    v8bf c0 = *(const v8bf*)(p);
    v8bf c1 = *(const v8bf*)(p + 16);
    return __builtin_shufflevector(c0, c1, 0,1,2,3,4,5,6,7,8,9,10,11,12,13,14,15);
}

// B tile (K=32 x N=16) from row-major [N][K] source (computes X * M^T):
// halves i<8 -> K=i+hi*8, i>=8 -> K=(i-8)+16+hi*8 (matches A layout).
__device__ __forceinline__ v16bf load_bt16(const __bf16* M, int ldm, int lane) {
    int n = lane & 15, hi = (lane >> 4) & 1;
    const __bf16* p = M + (size_t)n * ldm + hi * 8;
    v8bf c0 = *(const v8bf*)(p);
    v8bf c1 = *(const v8bf*)(p + 16);
    return __builtin_shufflevector(c0, c1, 0,1,2,3,4,5,6,7,8,9,10,11,12,13,14,15);
}

__device__ __forceinline__ float wred_max(float v) {
#pragma unroll
    for (int off = 16; off > 0; off >>= 1) v = fmaxf(v, __shfl_xor(v, off, 32));
    return v;
}
__device__ __forceinline__ float wred_sum(float v) {
#pragma unroll
    for (int off = 16; off > 0; off >>= 1) v += __shfl_xor(v, off, 32);
    return v;
}
__device__ __forceinline__ float hred_max(float v) {
#pragma unroll
    for (int off = 1; off < 16; off <<= 1) v = fmaxf(v, __shfl_xor(v, off, 32));
    return v;
}
__device__ __forceinline__ float hred_sum(float v) {
#pragma unroll
    for (int off = 1; off < 16; off <<= 1) v += __shfl_xor(v, off, 32);
    return v;
}

// CDNA5 async copy: global -> LDS, 16B per lane, tracked by ASYNCcnt (GVS mode)
__device__ __forceinline__ void async_b128(unsigned lds_off, unsigned voff, const void* sbase) {
    asm volatile("global_load_async_to_lds_b128 %0, %1, %2"
                 :: "v"(lds_off), "v"(voff), "s"(sbase) : "memory");
}

// ---------------- prep kernels ----------------

__global__ void k_cvt_emb(const float* __restrict__ emb, __bf16* __restrict__ embb) {
    size_t total = (size_t)VP * HP;
    for (size_t i = (size_t)blockIdx.x * blockDim.x + threadIdx.x; i < total;
         i += (size_t)gridDim.x * blockDim.x) {
        int r = (int)(i / HP), c = (int)(i % HP);
        float v = (r < V_ && c < H_) ? emb[(size_t)r * H_ + c] : 0.f;
        embb[i] = f2bf(v);
    }
}

__global__ void k_cvt_w(const float* __restrict__ wih, const float* __restrict__ whh,
                        __bf16* __restrict__ wihb, __bf16* __restrict__ whhb) {
    int total = 2 * 3 * H_ * HP;
    for (int i = blockIdx.x * blockDim.x + threadIdx.x; i < total;
         i += gridDim.x * blockDim.x) {
        int which = i / (3 * H_ * HP), j = i % (3 * H_ * HP);
        int r = j / HP, c = j % HP;
        const float* src = which ? whh : wih;
        __bf16* dst = which ? whhb : wihb;
        dst[j] = f2bf(c < H_ ? src[r * H_ + c] : 0.f);
    }
}

__global__ void k_init(const float* __restrict__ slot, const float* __restrict__ enc,
                       float* xf, __bf16* xb, float* hAf, __bf16* hAb, __bf16* hBb) {
    int total = R_ * HP;
    for (int i = blockIdx.x * blockDim.x + threadIdx.x; i < total;
         i += gridDim.x * blockDim.x) {
        int row = i / HP, c = i % HP;
        int s = row / B_, b = row % B_;
        float xv = (c < H_) ? slot[s * H_ + c] : 0.f;
        float hv = (c < H_) ? enc[b * H_ + c] : 0.f;
        xb[i] = f2bf(xv);
        hAb[i] = f2bf(hv);
        if (c >= H_) hBb[i] = f2bf(0.f);
        if (c < H_) {
            xf[(size_t)row * H_ + c] = xv;
            hAf[(size_t)row * H_ + c] = hv;
        }
    }
}

// ---------------- GRU cell (WMMA bf16) ----------------
__global__ __launch_bounds__(32) void k_gru(
    const __bf16* __restrict__ xb, const __bf16* __restrict__ hb,
    const float* __restrict__ hf,
    const __bf16* __restrict__ wih, const __bf16* __restrict__ whh,
    const float* __restrict__ bih, const float* __restrict__ bhh,
    float* __restrict__ hnf, __bf16* __restrict__ hnb) {
    int lane = threadIdx.x & 31;
    int rt = blockIdx.x, jt = blockIdx.y;
    v8f axr = {}, axz = {}, axn = {}, ahr = {}, ahz = {}, ahn = {};
    const __bf16* X = xb + (size_t)rt * 16 * HP;
    const __bf16* Hh = hb + (size_t)rt * 16 * HP;
    for (int k0 = 0; k0 < HP; k0 += 32) {
        v16bf a_x = load_a16(X + k0, HP, lane);
        v16bf a_h = load_a16(Hh + k0, HP, lane);
        v16bf w0 = load_bt16(wih + (size_t)(jt * 16) * HP + k0, HP, lane);
        v16bf w1 = load_bt16(wih + (size_t)(H_ + jt * 16) * HP + k0, HP, lane);
        v16bf w2 = load_bt16(wih + (size_t)(2 * H_ + jt * 16) * HP + k0, HP, lane);
        v16bf u0 = load_bt16(whh + (size_t)(jt * 16) * HP + k0, HP, lane);
        v16bf u1 = load_bt16(whh + (size_t)(H_ + jt * 16) * HP + k0, HP, lane);
        v16bf u2 = load_bt16(whh + (size_t)(2 * H_ + jt * 16) * HP + k0, HP, lane);
        __builtin_prefetch(wih + (size_t)(jt * 16) * HP + k0 + 32, 0, 3);
        axr = wmma_bf16(a_x, w0, axr);
        axz = wmma_bf16(a_x, w1, axz);
        axn = wmma_bf16(a_x, w2, axn);
        ahr = wmma_bf16(a_h, u0, ahr);
        ahz = wmma_bf16(a_h, u1, ahz);
        ahn = wmma_bf16(a_h, u2, ahn);
    }
    int n = lane & 15, hi = (lane >> 4) & 1;
    int col = jt * 16 + n;
    float bri = bih[col], bzi = bih[H_ + col], bni = bih[2 * H_ + col];
    float brh = bhh[col], bzh = bhh[H_ + col], bnh = bhh[2 * H_ + col];
#pragma unroll
    for (int r = 0; r < 8; r++) {
        int row = rt * 16 + r + hi * 8;
        float rg = 1.f / (1.f + __expf(-((axr[r] + bri) + (ahr[r] + brh))));
        float zg = 1.f / (1.f + __expf(-((axz[r] + bzi) + (ahz[r] + bzh))));
        float ng = tanhf((axn[r] + bni) + rg * (ahn[r] + bnh));
        float hold = hf[(size_t)row * H_ + col];
        float hv = (1.f - zg) * ng + zg * hold;
        hnf[(size_t)row * H_ + col] = hv;
        hnb[(size_t)row * HP + col] = f2bf(hv);
    }
}

// ---------------- attention + ratio + gate (one warp per (s,b) row) ----------------
__global__ __launch_bounds__(128) void k_att(
    const float* __restrict__ hn, const float* __restrict__ xf,
    const float* __restrict__ utt, const float* __restrict__ st,
    const int* __restrict__ ulen, const int* __restrict__ slen,
    const float* __restrict__ Wr, const float* __restrict__ brr,
    const float* __restrict__ Wg, const float* __restrict__ bg,
    float* __restrict__ cprob, float* __restrict__ sprob, float* __restrict__ sw,
    float* __restrict__ out, int t) {
    __shared__ float s_h[4][H_], s_c[4][H_], s_v[4][H_];
    __shared__ float s_pu[4][LU_], s_ps[4][LS_];
    int w = threadIdx.x >> 5, lane = threadIdx.x & 31;
    int row = blockIdx.x * 4 + w;
    int b = row % B_;
    for (int i = lane; i < H_; i += 32) s_h[w][i] = hn[(size_t)row * H_ + i];
    __syncthreads();
    int ul = ulen[b], sl = slen[b];

    float sc[8];
#pragma unroll
    for (int li = 0; li < 8; li++) {
        int l = lane + li * 32;
        const float4* p = (const float4*)(utt + ((size_t)b * LU_ + l) * H_);
        const float4* hh = (const float4*)s_h[w];
        float acc = 0.f;
        for (int k = 0; k < H_ / 4; k++) {
            float4 u = p[k], hq = hh[k];
            acc += u.x * hq.x + u.y * hq.y + u.z * hq.z + u.w * hq.w;
        }
        sc[li] = (l < ul) ? acc : NEG_INF;
    }
    float m = sc[0];
#pragma unroll
    for (int li = 1; li < 8; li++) m = fmaxf(m, sc[li]);
    m = wred_max(m);
    float ssum = 0.f;
#pragma unroll
    for (int li = 0; li < 8; li++) ssum += __expf(sc[li] - m);
    ssum = wred_sum(ssum);
#pragma unroll
    for (int li = 0; li < 8; li++) {
        int l = lane + li * 32;
        float p = __expf(sc[li] - m) / ssum;
        s_pu[w][l] = p;
        cprob[(size_t)row * LU_ + l] = p;
    }

    float sc2[2];
#pragma unroll
    for (int li = 0; li < 2; li++) {
        int l = lane + li * 32;
        const float4* p = (const float4*)(st + ((size_t)b * LS_ + l) * H_);
        const float4* hh = (const float4*)s_h[w];
        float acc = 0.f;
        for (int k = 0; k < H_ / 4; k++) {
            float4 u = p[k], hq = hh[k];
            acc += u.x * hq.x + u.y * hq.y + u.z * hq.z + u.w * hq.w;
        }
        sc2[li] = (l < sl) ? acc : NEG_INF;
    }
    float m2 = wred_max(fmaxf(sc2[0], sc2[1]));
    float ssum2 = wred_sum(__expf(sc2[0] - m2) + __expf(sc2[1] - m2));
#pragma unroll
    for (int li = 0; li < 2; li++) {
        int l = lane + li * 32;
        float p = __expf(sc2[li] - m2) / ssum2;
        s_ps[w][l] = p;
        sprob[(size_t)row * LS_ + l] = p;
    }
    __syncthreads();

    for (int h = lane; h < H_; h += 32) {
        float a = 0.f;
        for (int l = 0; l < LU_; l++) a += s_pu[w][l] * utt[((size_t)b * LU_ + l) * H_ + h];
        s_c[w][h] = a;
        float a2 = 0.f;
        for (int l = 0; l < LS_; l++) a2 += s_ps[w][l] * st[((size_t)b * LS_ + l) * H_ + h];
        s_v[w][h] = a2;
    }
    __syncthreads();

    float lr0 = 0.f, lr1 = 0.f, lr2 = 0.f;
    for (int f = lane; f < 4 * H_; f += 32) {
        float val = (f < H_) ? s_h[w][f]
                  : (f < 2 * H_) ? s_c[w][f - H_]
                  : (f < 3 * H_) ? s_v[w][f - 2 * H_]
                  : xf[(size_t)row * H_ + (f - 3 * H_)];
        lr0 += val * Wr[0 * 4 * H_ + f];
        lr1 += val * Wr[1 * 4 * H_ + f];
        lr2 += val * Wr[2 * 4 * H_ + f];
    }
    lr0 = wred_sum(lr0) + brr[0];
    lr1 = wred_sum(lr1) + brr[1];
    lr2 = wred_sum(lr2) + brr[2];
    float mm = fmaxf(lr0, fmaxf(lr1, lr2));
    float e0 = __expf(lr0 - mm), e1 = __expf(lr1 - mm), e2 = __expf(lr2 - mm);
    float es = e0 + e1 + e2;
    if (lane == 0) {
        sw[row * 3 + 0] = e0 / es;
        sw[row * 3 + 1] = e1 / es;
        sw[row * 3 + 2] = e2 / es;
    }

    if (t == 0) {
#pragma unroll
        for (int g = 0; g < NG_; g++) {
            float a = 0.f;
            for (int h = lane; h < H_; h += 32) a += s_c[w][h] * Wg[g * H_ + h];
            a = wred_sum(a);
            if (lane == 0)
                out[(size_t)S_ * B_ * T_ * V_ + (size_t)row * NG_ + g] = a + bg[g];
        }
    }
}

// ---------------- vocab logits: TDM-staged A, async double-buffered B ----------------
// grid (30 row-tiles, 157 vocab blocks of 128 cols), block = 4 waves, 2 col-tiles/wave
__global__ __launch_bounds__(128) void k_vocab(
    const __bf16* __restrict__ hnb, const __bf16* __restrict__ embb,
    float* __restrict__ out, float* __restrict__ pmax, float* __restrict__ psum, int t) {
    extern __shared__ char dynlds[];
    float (*lm)[16] = (float (*)[16])(dynlds + LM_OFF);
    float (*ls)[16] = (float (*)[16])(dynlds + LSM_OFF);
    int wv = threadIdx.x >> 5, lane = threadIdx.x & 31;
    int rt = blockIdx.x, vb = blockIdx.y;
    int n = lane & 15, hi = (lane >> 4) & 1;
    int cbase = vb * 128;

    // ---- stage A tile (16 x HP bf16) via Tensor Data Mover (wave 0 only) ----
    if (threadIdx.x < 32) {
        unsigned long long ga = (unsigned long long)(hnb + (size_t)rt * 16 * HP);
        u32x4 g0;
        g0[0] = 1u;                                   // count=1, user mode
        g0[1] = A_OFF;                                // lds_addr
        g0[2] = (unsigned)(ga & 0xffffffffu);         // global_addr[31:0]
        g0[3] = (unsigned)((ga >> 32) & 0x01ffffffu) | (2u << 30);  // addr[56:32] | type=2
        u32x8 g1;
        g1[0] = 0u;                                   // wg_mask=0, data_size=1B
        g1[1] = (unsigned)((HP * 2) & 0xffff) << 16;  // tensor_dim0 = 832 B (lo16)
        g1[2] = (unsigned)((R_ * 0 + 480) & 0xffff) << 16; // tensor_dim1 = 480 rows (lo16)
        g1[3] = (unsigned)((HP * 2) & 0xffff) << 16;  // tile_dim0 = 832 B
        g1[4] = 16u;                                  // tile_dim1 = 16 rows, tile_dim2 = 0
        g1[5] = (unsigned)(HP * 2);                   // tensor_dim0_stride = 832 (lo32)
        g1[6] = 0u;
        g1[7] = 0u;
        asm volatile("tensor_load_to_lds %0, %1" :: "s"(g0), "s"(g1) : "memory");
    }

    // ---- async-prime B chunk 0 (128 rows x 64 B) ----
    {
        int r0 = lane >> 2, cb = (lane & 3) * 16;
        unsigned lds0 = B0_OFF + (unsigned)(wv * 2048 + lane * 16);
        unsigned voff = (unsigned)(((cbase + wv * 32 + r0) * HP) * 2 + cb);
        async_b128(lds0, voff, embb);
        async_b128(lds0 + 512u, voff + (unsigned)(8 * HP * 2), embb);
    }
    if (threadIdx.x < 32) __builtin_amdgcn_s_wait_tensorcnt(0);

    v8f acc0 = {}, acc1 = {};
    const __bf16* Albs = (const __bf16*)(dynlds + A_OFF);
#pragma unroll
    for (int kk = 0; kk < KT; kk++) {
        if (kk < KT - 1) {   // issue next chunk into the other buffer
            int r0 = lane >> 2, cb = (lane & 3) * 16;
            unsigned lds0 = ((kk + 1) & 1 ? B1_OFF : B0_OFF) + (unsigned)(wv * 2048 + lane * 16);
            unsigned voff = (unsigned)(((cbase + wv * 32 + r0) * HP) * 2 + (kk + 1) * 64 + cb);
            async_b128(lds0, voff, embb);
            async_b128(lds0 + 512u, voff + (unsigned)(8 * HP * 2), embb);
            asm volatile("s_wait_asynccnt 2" ::: "memory");   // current chunk landed
        } else {
            asm volatile("s_wait_asynccnt 0" ::: "memory");
        }
        __syncthreads();     // publish LDS fills from all waves (and A on first iter)
        const __bf16* Bbuf = (const __bf16*)(dynlds + ((kk & 1) ? B1_OFF : B0_OFF));
        v16bf a = load_a16(Albs + kk * 32, HP, lane);
        v16bf b0 = load_bt16(Bbuf + (size_t)(wv * 32) * 32, 32, lane);
        v16bf b1 = load_bt16(Bbuf + (size_t)(wv * 32 + 16) * 32, 32, lane);
        acc0 = wmma_bf16(a, b0, acc0);
        acc1 = wmma_bf16(a, b1, acc1);
        __syncthreads();     // done reading before this buffer is refilled
    }

    int c0 = cbase + wv * 32 + n, c1 = cbase + wv * 32 + 16 + n;
    bool ok0 = c0 < V_, ok1 = c1 < V_;
    float lmax[8], lsum[8];
#pragma unroll
    for (int r = 0; r < 8; r++) {
        float v0 = ok0 ? acc0[r] : NEG_INF;
        float v1 = ok1 ? acc1[r] : NEG_INF;
        float mrow = hred_max(fmaxf(v0, v1));
        float e = (ok0 ? __expf(acc0[r] - mrow) : 0.f) + (ok1 ? __expf(acc1[r] - mrow) : 0.f);
        e = hred_sum(e);
        lmax[r] = mrow;
        lsum[r] = e;
        int row = rt * 16 + r + hi * 8;
        size_t base = ((size_t)row * T_ + t) * V_;
        if (ok0) out[base + c0] = acc0[r];
        if (ok1) out[base + c1] = acc1[r];
    }
    if (n == 0) {
#pragma unroll
        for (int r = 0; r < 8; r++) {
            lm[wv][r + hi * 8] = lmax[r];
            ls[wv][r + hi * 8] = lsum[r];
        }
    }
    __syncthreads();
    if (threadIdx.x < 16) {
        int rr = threadIdx.x;
        float M = NEG_INF;
#pragma unroll
        for (int w2 = 0; w2 < 4; w2++) M = fmaxf(M, lm[w2][rr]);
        float Ssum = 0.f;
#pragma unroll
        for (int w2 = 0; w2 < 4; w2++) {
            float e = ls[w2][rr];
            if (e > 0.f) Ssum += e * __expf(lm[w2][rr] - M);
        }
        int row = rt * 16 + rr;
        pmax[(size_t)row * NVB + vb] = M;
        psum[(size_t)row * NVB + vb] = Ssum;
    }
}

__global__ void k_reduce(const float* __restrict__ pmax, const float* __restrict__ psum,
                         float* __restrict__ rowM, float* __restrict__ rowS) {
    int row = blockIdx.x * blockDim.x + threadIdx.x;
    if (row >= R_) return;
    float M = NEG_INF;
    for (int vb = 0; vb < NVB; vb++) M = fmaxf(M, pmax[(size_t)row * NVB + vb]);
    float Ssum = 0.f;
    for (int vb = 0; vb < NVB; vb++) {
        float e = psum[(size_t)row * NVB + vb];
        if (e > 0.f) Ssum += e * __expf(pmax[(size_t)row * NVB + vb] - M);
    }
    rowM[row] = M;
    rowS[row] = Ssum;
}

__global__ void k_final(float* __restrict__ out, const float* __restrict__ rowM,
                        const float* __restrict__ rowS, const float* __restrict__ sw, int t) {
    int idx = blockIdx.x * blockDim.x + threadIdx.x;
    if (idx >= R_ * V_) return;
    int row = idx / V_, v = idx - row * V_;
    size_t o = ((size_t)row * T_ + t) * V_ + v;
    float p = __expf(out[o] - rowM[row]) / rowS[row];
    out[o] = sw[row * 3 + 0] * p;
}

__global__ void k_scatter(float* __restrict__ out, const float* __restrict__ cprob,
                          const float* __restrict__ sprob, const float* __restrict__ sw,
                          const int* __restrict__ dlg, const int* __restrict__ lst, int t) {
    int idx = blockIdx.x * blockDim.x + threadIdx.x;
    int totU = R_ * LU_;
    if (idx < totU) {
        int row = idx / LU_, l = idx - row * LU_;
        int b = row % B_;
        float wgt = 1.f - sw[row * 3 + 0] - sw[row * 3 + 1];
        int tok = dlg[b * LU_ + l];
        atomicAdd(out + ((size_t)row * T_ + t) * V_ + tok, wgt * cprob[idx]);
    } else if (idx < totU + R_ * LS_) {
        int j = idx - totU;
        int row = j / LS_, l = j - row * LS_;
        int b = row % B_;
        float wgt = sw[row * 3 + 1];
        int tok = lst[b * LS_ + l];
        atomicAdd(out + ((size_t)row * T_ + t) * V_ + tok, wgt * sprob[j]);
    }
}

__global__ void k_next_x(const float* __restrict__ emb, const int* __restrict__ target,
                         float* __restrict__ xf, __bf16* __restrict__ xb, int t) {
    int total = R_ * HP;
    for (int i = blockIdx.x * blockDim.x + threadIdx.x; i < total;
         i += gridDim.x * blockDim.x) {
        int row = i / HP, c = i % HP;
        int s = row / B_, b = row % B_;
        int tok = target[(b * S_ + s) * T_ + t];
        float v = (c < H_) ? emb[(size_t)tok * H_ + c] : 0.f;
        xb[i] = f2bf(v);
        if (c < H_) xf[(size_t)row * H_ + c] = v;
    }
}

extern "C" void kernel_launch(void* const* d_in, const int* in_sizes, int n_in,
                              void* d_out, int out_size, void* d_ws, size_t ws_size,
                              hipStream_t stream) {
    (void)in_sizes; (void)n_in; (void)out_size; (void)ws_size;
    const float* utt  = (const float*)d_in[0];
    const float* st   = (const float*)d_in[1];
    const float* enc  = (const float*)d_in[2];
    const float* slot = (const float*)d_in[3];
    const float* emb  = (const float*)d_in[4];
    const float* wih  = (const float*)d_in[5];
    const float* whh  = (const float*)d_in[6];
    const float* bih  = (const float*)d_in[7];
    const float* bhh  = (const float*)d_in[8];
    const float* Wr   = (const float*)d_in[9];
    const float* brr  = (const float*)d_in[10];
    const float* Wg   = (const float*)d_in[11];
    const float* bg   = (const float*)d_in[12];
    const int* target = (const int*)d_in[13];
    const int* dlg    = (const int*)d_in[14];
    const int* lst    = (const int*)d_in[15];
    const int* ulen   = (const int*)d_in[16];
    const int* slen   = (const int*)d_in[17];
    float* out = (float*)d_out;

    char* w = (char*)d_ws;
    size_t off = 0;
#define TAKE(T, name, bytes) \
    T name = (T)(w + off); off = (off + (size_t)(bytes) + 255) & ~(size_t)255;
    TAKE(__bf16*, embb,  (size_t)VP * HP * 2)
    TAKE(__bf16*, wihb,  (size_t)3 * H_ * HP * 2)
    TAKE(__bf16*, whhb,  (size_t)3 * H_ * HP * 2)
    TAKE(__bf16*, xb,    (size_t)R_ * HP * 2)
    TAKE(float*,  xf,    (size_t)R_ * H_ * 4)
    TAKE(float*,  hAf,   (size_t)R_ * H_ * 4)
    TAKE(float*,  hBf,   (size_t)R_ * H_ * 4)
    TAKE(__bf16*, hAb,   (size_t)R_ * HP * 2)
    TAKE(__bf16*, hBb,   (size_t)R_ * HP * 2)
    TAKE(float*,  cprob, (size_t)R_ * LU_ * 4)
    TAKE(float*,  sprob, (size_t)R_ * LS_ * 4)
    TAKE(float*,  swb,   (size_t)R_ * 3 * 4)
    TAKE(float*,  pmax,  (size_t)R_ * NVB * 4)
    TAKE(float*,  psum,  (size_t)R_ * NVB * 4)
    TAKE(float*,  rowM,  (size_t)R_ * 4)
    TAKE(float*,  rowS,  (size_t)R_ * 4)
#undef TAKE

    k_cvt_emb<<<8192, 256, 0, stream>>>(emb, embb);
    k_cvt_w<<<1024, 256, 0, stream>>>(wih, whh, wihb, whhb);
    k_init<<<780, 256, 0, stream>>>(slot, enc, xf, xb, hAf, hAb, hBb);

    for (int t = 0; t < T_; t++) {
        const float*  hpf = (t & 1) ? hBf : hAf;
        const __bf16* hpb = (t & 1) ? hBb : hAb;
        float*  hnf = (t & 1) ? hAf : hBf;
        __bf16* hnb = (t & 1) ? hAb : hBb;

        k_gru<<<dim3(30, 25), 32, 0, stream>>>(xb, hpb, hpf, wihb, whhb, bih, bhh, hnf, hnb);
        k_att<<<120, 128, 0, stream>>>(hnf, xf, utt, st, ulen, slen, Wr, brr, Wg, bg,
                                       cprob, sprob, swb, out, t);
        k_vocab<<<dim3(30, NVB), 128, VOC_LDS, stream>>>(hnb, embb, out, pmax, psum, t);
        k_reduce<<<(R_ + 255) / 256, 256, 0, stream>>>(pmax, psum, rowM, rowS);
        k_final<<<(R_ * V_ + 255) / 256, 256, 0, stream>>>(out, rowM, rowS, swb, t);
        k_scatter<<<(R_ * (LU_ + LS_) + 255) / 256, 256, 0, stream>>>(out, cprob, sprob,
                                                                     swb, dlg, lst, t);
        if (t < T_ - 1)
            k_next_x<<<780, 256, 0, stream>>>(emb, target, xf, xb, t);
    }
}
